// LaplaceTransformEncoder_65429531787732
// MI455X (gfx1250) — matrix operand order repaired and checked by reading, not compiled, not run on hardware
//
#include <hip/hip_runtime.h>
#include <math.h>

// ---- problem constants (match reference) ----
#define BATCH   64
#define TLEN    4096
#define FEAT    64
#define CONDD   128
#define KP      64      // K poles
#define TWOK    128     // 2K
#define OMEGA_MAXF 3.14159265358979323846f
#define ALPHA_MINF 1e-6f
#define BASE_LAMBDA 1e-3f

// ---- vector types ----
typedef _Float16 v16h __attribute__((ext_vector_type(16)));
typedef _Float16 v8h  __attribute__((ext_vector_type(8)));
typedef _Float16 v2h  __attribute__((ext_vector_type(2)));
typedef float    v8f  __attribute__((ext_vector_type(8)));
typedef unsigned int u32x4 __attribute__((ext_vector_type(4)));
typedef int          i32x8 __attribute__((ext_vector_type(8)));
typedef int          i32x4 __attribute__((ext_vector_type(4)));

// ---- LDS layout (dynamic shared, regions overlap across phases) ----
// GEMM phase: two complete buffer sets (double buffered)
#define LDA 72                                   // f16 row stride for ldsA (128 rows)
#define LDX 72                                   // f16 row stride for ldsX (64 rows)
#define OFF_X_IN  (TWOK * LDA * 2)               // 18432 within a buffer set
#define OFF_S_IN  (OFF_X_IN + FEAT * LDX * 2)    // 27648 within a buffer set
#define GEMMBUF   (OFF_S_IN + 64 * FEAT * 4)     // 44032 bytes per set (x2 = 88064)
// Solver phase (reuses the same LDS):
#define SG 129                                   // f32 row stride for G
#define SR 65                                    // f32 row stride for R
#define OFF_G   0
#define OFF_R   (OFF_G + TWOK * SG * 4)          // 66048
#define SMEM_BYTES (OFF_R + TWOK * SR * 4)       // 99328 (>= 2*GEMMBUF)

__device__ __forceinline__ float softplusf_(float v) {
    return (v > 20.f) ? v : log1pf(__expf(v));
}
__device__ __forceinline__ float sigmoidf_(float v) {
    return 1.f / (1.f + __expf(-v));
}
__device__ __forceinline__ v16h load_frag(const _Float16* p0, const _Float16* p1) {
    v8h lo = *(const v8h*)p0;
    v8h hi = *(const v8h*)p1;
    return __builtin_shufflevector(lo, hi, 0,1,2,3,4,5,6,7,8,9,10,11,12,13,14,15);
}

// Issue a TDM DMA of a 64x64 f32 tile (row stride FEAT) into LDS at ldsOff.
__device__ __forceinline__ void tdm_load_tile(const float* gptr, unsigned ldsOff) {
    unsigned long long ga = (unsigned long long)(uintptr_t)gptr;
    u32x4 g0;
    g0[0] = 1u;                                     // count=1, user mode
    g0[1] = ldsOff;                                 // lds_addr (bytes)
    g0[2] = (unsigned)(ga & 0xffffffffull);         // global_addr[31:0]
    g0[3] = (unsigned)((ga >> 32) & 0x01ffffffull)  // global_addr[56:32]
          | 0x80000000u;                            // type=2 ("image")
    i32x8 g1;
    g1[0] = (2 << 16);                              // data_size = 4 bytes
    g1[1] = (FEAT & 0xffff) << 16;                  // tensor_dim0 lo16
    g1[2] = (FEAT >> 16) | ((64 & 0xffff) << 16);   // dim0 hi16 | tensor_dim1 lo16
    g1[3] = (64 >> 16) | (FEAT << 16);              // dim1 hi16 | tile_dim0
    g1[4] = 64;                                     // tile_dim1 (tile_dim2=0)
    g1[5] = FEAT;                                   // tensor_dim0_stride lo32
    g1[6] = 0;                                      // stride0 hi16 | stride1 lo16
    g1[7] = 0;                                      // stride1 hi32
    i32x4 g2 = {0, 0, 0, 0};
    i32x4 g3 = {0, 0, 0, 0};
    i32x8 g4 = {0, 0, 0, 0, 0, 0, 0, 0};            // unused group (2D tensor)
    __builtin_amdgcn_tensor_load_to_lds(g0, g1, g2, g3, g4, 0);
}

__global__ __launch_bounds__(256, 1)
void laplace_encoder_kernel(const float* __restrict__ x,
                            const float* __restrict__ cond,
                            const float* __restrict__ rho_raw,
                            const float* __restrict__ omega_raw,
                            const float* __restrict__ Wp,
                            const float* __restrict__ bp,
                            const float* __restrict__ Wl,
                            const float* __restrict__ bl,
                            float* __restrict__ out)
{
    extern __shared__ unsigned char smem[];
    __shared__ float sh_h[CONDD];
    __shared__ float sh_rho[KP];
    __shared__ float sh_omega[KP];
    __shared__ float sh_lam;

    const int b    = blockIdx.x;
    const int tid  = threadIdx.x;
    const int lane = tid & 31;
    const int wave = tid >> 5;
    const int m16  = wave * 16;          // this wave's M-tile base row

    // ---------------- phase 0: conditional pole parameters ----------------
    if (tid < CONDD) {
        float c = cond[b * CONDD + tid];
        sh_h[tid] = c * sigmoidf_(c);            // SiLU
    }
    __syncthreads();
    if (tid < TWOK) {
        float acc = bp[tid];
        #pragma unroll 4
        for (int i = 0; i < CONDD; ++i) acc += sh_h[i] * Wp[i * TWOK + tid];
        if (tid < KP) {
            float rho0 = softplusf_(rho_raw[tid]) + ALPHA_MINF;
            float drho = 0.5f * tanhf(acc);
            sh_rho[tid] = softplusf_(rho0 + drho) + ALPHA_MINF;
        } else {
            int k = tid - KP;
            float om0   = OMEGA_MAXF * sigmoidf_(omega_raw[k]);
            float p0    = fminf(fmaxf(om0 / OMEGA_MAXF, 1e-4f), 1.f - 1e-4f);
            float lgt0  = logf(p0) - log1pf(-p0);
            float domg  = 0.5f * tanhf(acc);
            sh_omega[k] = OMEGA_MAXF * sigmoidf_(lgt0 + domg);
        }
    }
    if (tid == 0) {
        float s = bl[0];
        for (int i = 0; i < CONDD; ++i) s += sh_h[i] * Wl[i];
        sh_lam = BASE_LAMBDA * (5.f * sigmoidf_(s) + 0.5f);
    }
    __syncthreads();

    // ---------------- phase 1: fused GEMM  C = Bmat^T [Bmat | x] ----------------
    v8f acc[12];
    #pragma unroll
    for (int j = 0; j < 12; ++j)
        #pragma unroll
        for (int r = 0; r < 8; ++r) acc[j][r] = 0.f;

    const unsigned stgLds0 = (unsigned)(uintptr_t)(void*)(smem + OFF_S_IN);
    const unsigned stgLds1 = (unsigned)(uintptr_t)(void*)(smem + GEMMBUF + OFF_S_IN);
    const float* xb = x + (size_t)b * TLEN * FEAT;

    const int NT = TLEN / 64;
    // prologue: DMA tile 0 into staging buffer 0
    if (tid == 0) tdm_load_tile(xb, stgLds0);

    for (int it = 0; it < NT; ++it) {
        const int p  = it & 1;
        const int t0 = it * 64;
        _Float16* ldsA = (_Float16*)(smem + p * GEMMBUF);
        _Float16* ldsX = (_Float16*)(smem + p * GEMMBUF + OFF_X_IN);
        float*    stg  = (float*)   (smem + p * GEMMBUF + OFF_S_IN);

        // wave0: make sure TDM for tile `it` has landed (only one in flight here)
        if (tid == 0) __builtin_amdgcn_s_wait_tensorcnt(0);
        __syncthreads();   // staging visible; buf p free of last-but-one tile's readers

        // ---- generate basis tile (transposed), two time steps per b32 store ----
        for (int idx = tid; idx < 32 * KP; idx += 256) {
            int k  = idx & 63;
            int tl = (idx >> 6) * 2;
            float tA = (float)(t0 + tl), tB = tA + 1.f;
            float rho = sh_rho[k], om = sh_omega[k];
            float eA = __expf(-tA * rho), eB = __expf(-tB * rho);
            float sA, cA, sB, cB;
            __sincosf(tA * om, &sA, &cA);
            __sincosf(tB * om, &sB, &cB);
            v2h pc = { (_Float16)(eA * cA), (_Float16)(eB * cB) };
            v2h ps = { (_Float16)(eA * sA), (_Float16)(eB * sB) };
            *(v2h*)(ldsA + k        * LDA + tl) = pc;
            *(v2h*)(ldsA + (KP + k) * LDA + tl) = ps;
        }
        // ---- x: staging [tl][d] f32 -> ldsX[d][tl] f16, two per b32 store ----
        for (int idx = tid; idx < 32 * FEAT; idx += 256) {
            int d  = idx & 63;
            int tl = (idx >> 6) * 2;
            v2h px = { (_Float16)stg[tl * FEAT + d],
                       (_Float16)stg[(tl + 1) * FEAT + d] };
            *(v2h*)(ldsX + d * LDX + tl) = px;
        }
        // pipeline: kick off DMA for the next tile into the other staging buffer
        if (tid == 0 && it + 1 < NT)
            tdm_load_tile(xb + (size_t)(t0 + 64) * FEAT, p ? stgLds0 : stgLds1);
        __syncthreads();

        // ---- WMMA: two K=32 steps; B fragments double-buffered for overlap ----
        const int mrow = m16 + (lane & 15);
        const int hh   = lane >> 4;
        const int nlo  = lane & 15;
        #pragma unroll
        for (int kt = 0; kt < 64; kt += 32) {
            const _Float16* pa = ldsA + mrow * LDA + kt + 16 * hh;
            v16h afrag = load_frag(pa, pa + 8);
            v16h bbuf[2];
            {
                const _Float16* pb = ldsA + nlo * LDA + kt + 16 * hh;
                bbuf[0] = load_frag(pb, pb + 8);
            }
            #pragma unroll
            for (int j = 0; j < 12; ++j) {
                if (j < 11) {
                    int col = (j + 1) * 16 + nlo;
                    const _Float16* pb = (j + 1 < 8)
                        ? (ldsA + col * LDA + kt + 16 * hh)
                        : (ldsX + (col - TWOK) * LDX + kt + 16 * hh);
                    bbuf[(j + 1) & 1] = load_frag(pb, pb + 8);
                }
                acc[j] = __builtin_amdgcn_wmma_f32_16x16x32_f16(
                    false, afrag, false, bbuf[j & 1], (short)0, acc[j], false, false);
            }
        }
        // no trailing barrier: next iteration's generate writes the other buffer;
        // the barrier at the top of the next iteration provides the needed fence.
    }
    __syncthreads();

    // ---------------- phase 2: spill accumulators to LDS (G | R) ----------------
    float* G = (float*)(smem + OFF_G);
    float* R = (float*)(smem + OFF_R);
    {
        const int nlo = lane & 15;
        const int mhi = (lane >> 4) * 8;
        #pragma unroll
        for (int j = 0; j < 12; ++j)
            #pragma unroll
            for (int r = 0; r < 8; ++r) {
                int row = m16 + mhi + r;
                int col = j * 16 + nlo;
                float v = acc[j][r];
                if (j < 8) G[row * SG + col] = v;
                else       R[row * SR + (col - TWOK)] = v;
            }
    }
    __syncthreads();
    if (tid < TWOK) G[tid * SG + tid] += sh_lam;   // ridge
    __syncthreads();

    // ---------------- phase 3: Cholesky G = L L^T (in place, symmetric) -------
    for (int j = 0; j < TWOK; ++j) {
        if (tid == 0) G[j * SG + j] = sqrtf(G[j * SG + j]);
        __syncthreads();
        float d = G[j * SG + j];
        float inv = 1.f / d;
        for (int i = j + 1 + tid; i < TWOK; i += 256) G[i * SG + j] *= inv;
        __syncthreads();
        int n = TWOK - 1 - j;
        for (int idx = tid; idx < n * n; idx += 256) {
            int r = j + 1 + idx / n;
            int c = j + 1 + idx % n;
            G[r * SG + c] -= G[r * SG + j] * G[c * SG + j];
        }
        __syncthreads();
    }

    // ---------------- phase 4: forward solve L Y = BtX ----------------
    for (int j = 0; j < TWOK; ++j) {
        float inv = 1.f / G[j * SG + j];
        for (int d = tid; d < FEAT; d += 256) R[j * SR + d] *= inv;
        __syncthreads();
        int n = TWOK - 1 - j;
        for (int idx = tid; idx < n * FEAT; idx += 256) {
            int i = j + 1 + (idx >> 6);
            int d = idx & 63;
            R[i * SR + d] -= G[i * SG + j] * R[j * SR + d];
        }
        __syncthreads();
    }

    // ---------------- phase 5: backward solve L^T Theta = Y ----------------
    for (int j = TWOK - 1; j >= 0; --j) {
        float inv = 1.f / G[j * SG + j];
        for (int d = tid; d < FEAT; d += 256) R[j * SR + d] *= inv;
        __syncthreads();
        for (int idx = tid; idx < j * FEAT; idx += 256) {
            int i = idx >> 6;
            int d = idx & 63;
            R[i * SR + d] -= G[j * SG + i] * R[j * SR + d];   // L^T[i][j] = L[j][i]
        }
        __syncthreads();
    }

    // ---------------- phase 6: write Theta [B, 2K, FEAT] ----------------
    float* ob = out + (size_t)b * TWOK * FEAT;
    for (int idx = tid; idx < TWOK * FEAT; idx += 256) {
        int m = idx >> 6;
        int d = idx & 63;
        ob[idx] = R[m * SR + d];
    }
}

extern "C" void kernel_launch(void* const* d_in, const int* in_sizes, int n_in,
                              void* d_out, int out_size, void* d_ws, size_t ws_size,
                              hipStream_t stream) {
    (void)in_sizes; (void)n_in; (void)d_ws; (void)ws_size; (void)out_size;
    const float* x         = (const float*)d_in[0];
    const float* cond      = (const float*)d_in[1];
    const float* rho_raw   = (const float*)d_in[2];
    const float* omega_raw = (const float*)d_in[3];
    const float* Wp        = (const float*)d_in[4];
    const float* bp        = (const float*)d_in[5];
    const float* Wl        = (const float*)d_in[6];
    const float* bl        = (const float*)d_in[7];
    float* out = (float*)d_out;

    dim3 grid(BATCH), block(256);
    laplace_encoder_kernel<<<grid, block, SMEM_BYTES, stream>>>(
        x, cond, rho_raw, omega_raw, Wp, bp, Wl, bl, out);
}
